// DualAttention_48601849922254
// MI455X (gfx1250) — matrix operand order repaired and checked by reading, not compiled, not run on hardware
//
#include <hip/hip_runtime.h>
#include <hip/hip_bf16.h>

typedef unsigned short u16;
typedef unsigned long long u64;
typedef __attribute__((ext_vector_type(16))) __bf16 v16bf;
typedef __attribute__((ext_vector_type(8)))  float  v8f;
typedef __attribute__((ext_vector_type(4)))  unsigned int v4u;

union AFrag { v16bf bf; v4u u[2]; };
union Pack8 { u16 h[8]; v4u u; };

__device__ __forceinline__ u16 f2bf(float f) {
  unsigned u = __float_as_uint(f);
  u += 0x7FFFu + ((u >> 16) & 1u);   // round-to-nearest-even
  return (u16)(u >> 16);
}
__device__ __forceinline__ float bf2f(u16 h) {
  return __uint_as_float(((unsigned)h) << 16);
}

// CDNA5 async memory->LDS copy (16B per lane), tracked by ASYNCcnt
__device__ __forceinline__ void async_cp16(unsigned lds_addr, const u16* gptr) {
  u64 ga = (u64)(size_t)gptr;
  asm volatile("global_load_async_to_lds_b128 %0, %1, off"
               :: "v"(lds_addr), "v"(ga) : "memory");
}
__device__ __forceinline__ void wait_async0() {
  asm volatile("s_wait_asynccnt 0x0" ::: "memory");
}

// problem dims
constexpr int BB = 4;
constexpr int C  = 512;
constexpr int CQ = 64;
constexpr int NN = 4096;
constexpr int C2 = 1024;
constexpr float BN_EPS = 1e-5f;

// GEMM tile: block = 64(M) x 256(N), K step 32, 256 threads = 8 waves (2Mx4N),
// each wave computes 32x64 = 2 A-frags x 4 B-frags = 8 WMMAs / k-step.
constexpr int MT = 64, NT = 256, KT = 32;

// C = A[M,K] x Bt[N,K]^T.  A, Bt row-major bf16 in global.
// MODE: 0 = f32 store (+bias)
//       1 = bf16 store (+bias)
//       2 = bf16 TRANSPOSED store (+bias)            CT[n*ldc+m]
//       4 = bf16 TRANSPOSED store, scale*acc + res[m*ldres+n]
//       5 = bf16 store,            scale*acc + res[n*ldres+m]
template<int MODE>
__global__ __launch_bounds__(256) void gemm_bf16(
    const u16* __restrict__ A, size_t sA, int lda,
    const u16* __restrict__ Bt, size_t sB, int ldbt,
    void* __restrict__ Cm, size_t sC, int ldc,
    const float* __restrict__ bias,
    const float* __restrict__ scaleptr,
    const float* __restrict__ res, size_t sRes, int ldres,
    int K)
{
  __shared__ u16 As[2][MT * KT];   // [64][32] row-major, double buffered (8KB)
  __shared__ u16 Bs[2][NT * KT];   // [256][32] rows of B^T (32KB)

  const int tid = threadIdx.x;
  const int bz  = blockIdx.z;
  const int m0  = blockIdx.y * MT;
  const int n0  = blockIdx.x * NT;
  A  += bz * sA;
  Bt += bz * sB;

  const int wave = tid >> 5, lane = tid & 31;
  const int g  = lane >> 4;          // lane group
  const int lr = lane & 15;
  const int mw = (wave & 1) * 32;    // wave's M offset in tile (2 groups)
  const int nw = (wave >> 1) * 64;   // wave's N offset in tile (4 groups)

  // async tile-copy mapping: A = 1 chunk/thread; B = thread t copies B^T row t (4 chunks)
  const int arow = tid >> 2;         // 0..63
  const int acol = (tid & 3) * 8;    // 0,8,16,24

  auto issue_tile = [&](int buf, int k0) {
    async_cp16((unsigned)(size_t)&As[buf][arow * KT + acol],
               A + (size_t)(m0 + arow) * lda + (k0 + acol));
    const u16* brow = Bt + (size_t)(n0 + tid) * ldbt + k0;
    unsigned lrow = (unsigned)(size_t)&Bs[buf][tid * KT];
#pragma unroll
    for (int c = 0; c < 4; ++c) async_cp16(lrow + 16 * c, brow + 8 * c);
  };

  v8f acc[2][4] = {};

  const int nk = K / KT;
  issue_tile(0, 0);
  wait_async0();
  __syncthreads();

  for (int kt = 0; kt < nk; ++kt) {
    const int buf = kt & 1;
    if (kt + 1 < nk) issue_tile(buf ^ 1, (kt + 1) * KT);

    // A fragments (ISA 7.12.2): comps 0..7 <- K=8g..8g+7, comps 8..15 <- K=16+8g..
    AFrag a0, a1;
    a0.u[0] = *(const v4u*)&As[buf][(mw + lr) * KT + 8 * g];
    a0.u[1] = *(const v4u*)&As[buf][(mw + lr) * KT + 16 + 8 * g];
    a1.u[0] = *(const v4u*)&As[buf][(mw + 16 + lr) * KT + 8 * g];
    a1.u[1] = *(const v4u*)&As[buf][(mw + 16 + lr) * KT + 16 + 8 * g];
#pragma unroll
    for (int j = 0; j < 4; ++j) {
      // B fragment: lane = col, comps 0..7 <- K=8g..8g+7, comps 8..15 <- K=16+8g.. (rows of B^T are K-contiguous)
      AFrag b;
      const u16* bb = &Bs[buf][(nw + j * 16 + lr) * KT + 8 * g];
      b.u[0] = *(const v4u*)bb;
      b.u[1] = *(const v4u*)(bb + 16);
      acc[0][j] = __builtin_amdgcn_wmma_f32_16x16x32_bf16(
          false, a0.bf, false, b.bf, (short)0, acc[0][j], false, false);
      acc[1][j] = __builtin_amdgcn_wmma_f32_16x16x32_bf16(
          false, a1.bf, false, b.bf, (short)0, acc[1][j], false, false);
    }

    wait_async0();      // my next-tile copies have landed in LDS
    __syncthreads();    // everyone done computing buf + next tile visible
  }

  // epilogue: C/D layout: VGPR r, lane -> m = mi*16 + r + 8g, n = lr
  float bv[2][8];
#pragma unroll
  for (int mi = 0; mi < 2; ++mi)
#pragma unroll
    for (int r = 0; r < 8; ++r) bv[mi][r] = 0.f;
  if ((MODE == 0 || MODE == 1 || MODE == 2) && bias != nullptr) {
#pragma unroll
    for (int mi = 0; mi < 2; ++mi)
#pragma unroll
      for (int r = 0; r < 8; ++r)
        bv[mi][r] = bias[m0 + mw + mi * 16 + r + 8 * g];
  }
  float scale = 0.f;
  if (MODE == 4 || MODE == 5) scale = scaleptr[0];

#pragma unroll
  for (int mi = 0; mi < 2; ++mi) {
#pragma unroll
    for (int j = 0; j < 4; ++j) {
      const int ng = n0 + nw + j * 16 + lr;
      const int mbase = m0 + mw + mi * 16;
      if (MODE == 0) {
        float* Co = (float*)Cm + bz * sC;
#pragma unroll
        for (int r = 0; r < 8; ++r)
          Co[(size_t)(mbase + r + 8 * g) * ldc + ng] = acc[mi][j][r] + bv[mi][r];
      } else if (MODE == 1) {
        u16* Co = (u16*)Cm + bz * sC;
#pragma unroll
        for (int r = 0; r < 8; ++r)
          Co[(size_t)(mbase + r + 8 * g) * ldc + ng] = f2bf(acc[mi][j][r] + bv[mi][r]);
      } else if (MODE == 2) {
        u16* Co = (u16*)Cm + bz * sC;
        const int mstart = mbase + 8 * g;   // r=0..7 contiguous along m
        Pack8 p;
#pragma unroll
        for (int r = 0; r < 8; ++r) p.h[r] = f2bf(acc[mi][j][r] + bv[mi][r]);
        *(v4u*)&Co[(size_t)ng * ldc + mstart] = p.u;
      } else if (MODE == 4) {  // chan: CT[n, m] = bf16(scale*acc + res[m, n])
        u16* Co = (u16*)Cm + bz * sC;
        const float* R = res + bz * sRes;
        const int mstart = mbase + 8 * g;
        Pack8 p;
#pragma unroll
        for (int r = 0; r < 8; ++r) {
          float v = scale * acc[mi][j][r] + R[(size_t)(mstart + r) * ldres + ng];
          p.h[r] = f2bf(v);
        }
        *(v4u*)&Co[(size_t)ng * ldc + mstart] = p.u;
      } else {                 // MODE 5, pos: C[m, n] = bf16(scale*acc + res[n, m])
        u16* Co = (u16*)Cm + bz * sC;
        const float* R = res + bz * sRes;
#pragma unroll
        for (int r = 0; r < 8; ++r) {
          const int m = mbase + r + 8 * g;
          float v = scale * acc[mi][j][r] + R[(size_t)ng * ldres + m];
          Co[(size_t)m * ldc + ng] = f2bf(v);
        }
      }
    }
  }
}

__global__ __launch_bounds__(256) void cvt_bf16(const float* __restrict__ in,
                                                u16* __restrict__ out) {
  size_t i = (size_t)blockIdx.x * 256 + threadIdx.x;
  out[i] = f2bf(in[i]);
}

// xT[b][n][c] = x[b][c][n]
__global__ __launch_bounds__(256) void transpose_cvt(const float* __restrict__ x,
                                                     u16* __restrict__ xT) {
  size_t i = (size_t)blockIdx.x * 256 + threadIdx.x;
  size_t b = i / ((size_t)NN * C);
  size_t r = i % ((size_t)NN * C);
  size_t n = r / C, c = r % C;
  xT[i] = f2bf(x[(b * C + c) * NN + n]);
}

// in-place row softmax on bf16 matrix, one block per row, cols/256 <= 16
__global__ __launch_bounds__(256) void softmax_rows(u16* __restrict__ p, int cols) {
  __shared__ float sred[256];
  const int tid = threadIdx.x;
  u16* row = p + (size_t)blockIdx.x * cols;
  const int per = cols >> 8;
  float v[16];
  float m = -3.4e38f;
  for (int i = 0; i < per; ++i) {
    v[i] = bf2f(row[tid + (i << 8)]);
    m = fmaxf(m, v[i]);
  }
  sred[tid] = m; __syncthreads();
  for (int s = 128; s > 0; s >>= 1) {
    if (tid < s) sred[tid] = fmaxf(sred[tid], sred[tid + s]);
    __syncthreads();
  }
  m = sred[0]; __syncthreads();
  float sum = 0.f;
  for (int i = 0; i < per; ++i) { v[i] = __expf(v[i] - m); sum += v[i]; }
  sred[tid] = sum; __syncthreads();
  for (int s = 128; s > 0; s >>= 1) {
    if (tid < s) sred[tid] += sred[tid + s];
    __syncthreads();
  }
  float inv = 1.0f / sred[0];
  for (int i = 0; i < per; ++i) row[tid + (i << 8)] = f2bf(v[i] * inv);
}

// per-channel mean/var over (b, n); one block per channel
__global__ __launch_bounds__(256) void bn_stats(const float* __restrict__ y,
                                                float* __restrict__ stats) {
  __shared__ float s1[256], s2[256];
  const int c = blockIdx.x, tid = threadIdx.x;
  float s = 0.f, q = 0.f;
  for (int b = 0; b < BB; ++b) {
    const float* p = y + ((size_t)b * C + c) * NN;
    for (int i = tid; i < NN; i += 256) { float v = p[i]; s += v; q += v * v; }
  }
  s1[tid] = s; s2[tid] = q; __syncthreads();
  for (int st = 128; st > 0; st >>= 1) {
    if (tid < st) { s1[tid] += s1[tid + st]; s2[tid] += s2[tid + st]; }
    __syncthreads();
  }
  if (tid == 0) {
    float inv = 1.0f / (float)(BB * NN);
    float mean = s1[0] * inv;
    stats[c]     = mean;
    stats[C + c] = s2[0] * inv - mean * mean;
  }
}

__global__ __launch_bounds__(256) void bn_apply(float* __restrict__ y,
                                                const float* __restrict__ stats,
                                                const float* __restrict__ w,
                                                const float* __restrict__ b) {
  size_t i = (size_t)blockIdx.x * 256 + threadIdx.x;
  int c = (int)((i / NN) % C);
  float v = (y[i] - stats[c]) * rsqrtf(stats[C + c] + BN_EPS) * w[c] + b[c];
  y[i] = v > 0.f ? v : 0.f;
}

extern "C" void kernel_launch(void* const* d_in, const int* in_sizes, int n_in,
                              void* d_out, int out_size, void* d_ws, size_t ws_size,
                              hipStream_t stream) {
  (void)in_sizes; (void)n_in; (void)out_size; (void)ws_size;
  const float* x     = (const float*)d_in[0];
  const float* qw    = (const float*)d_in[1];
  const float* qb    = (const float*)d_in[2];
  const float* kw    = (const float*)d_in[3];
  const float* kb    = (const float*)d_in[4];
  const float* vw    = (const float*)d_in[5];
  const float* vb    = (const float*)d_in[6];
  const float* gamma = (const float*)d_in[7];
  const float* beta  = (const float*)d_in[8];
  const float* fw    = (const float*)d_in[9];
  const float* fb    = (const float*)d_in[10];
  const float* bn_w  = (const float*)d_in[11];
  const float* bn_b  = (const float*)d_in[12];
  float* out = (float*)d_out;

  const size_t CN  = (size_t)C * NN;
  const size_t NC  = (size_t)NN * C;
  const size_t NNl = (size_t)NN * NN;

  // workspace carve-up (bytes, 256B aligned)
  char* w = (char*)d_ws;
  size_t off = 0;
  auto take = [&](size_t bytes) { char* p = w + off; off = (off + bytes + 255) & ~(size_t)255; return p; };
  u16*   xbf    = (u16*)  take((size_t)BB * CN * 2);
  u16*   xT     = (u16*)  take((size_t)BB * NC * 2);
  u16*   qwb    = (u16*)  take((size_t)CQ * C * 2);
  u16*   kwb    = (u16*)  take((size_t)CQ * C * 2);
  u16*   vwb    = (u16*)  take((size_t)C * C * 2);
  u16*   fwb    = (u16*)  take((size_t)C * C2 * 2);
  u16*   qT     = (u16*)  take((size_t)BB * NN * CQ * 2);  // [N, CQ]
  u16*   kT     = (u16*)  take((size_t)BB * NN * CQ * 2);  // [N, CQ]
  u16*   vmat   = (u16*)  take((size_t)BB * CN * 2);       // [C, N]
  u16*   att    = (u16*)  take((size_t)BB * NNl * 2);      // 134 MB: L2-resident
  u16*   energy = (u16*)  take((size_t)BB * C * C * 2);
  u16*   catT   = (u16*)  take((size_t)BB * (size_t)NN * C2 * 2);  // [N, 2C]
  float* stats  = (float*)take((size_t)2 * C * 4);

  // --- bf16 conversions ---
  cvt_bf16<<<(unsigned)(BB * CN / 256), 256, 0, stream>>>(x, xbf);
  cvt_bf16<<<(unsigned)(CQ * C / 256), 256, 0, stream>>>(qw, qwb);
  cvt_bf16<<<(unsigned)(CQ * C / 256), 256, 0, stream>>>(kw, kwb);
  cvt_bf16<<<(unsigned)(C * C / 256), 256, 0, stream>>>(vw, vwb);
  cvt_bf16<<<(unsigned)(C * C2 / 256), 256, 0, stream>>>(fw, fwb);
  transpose_cvt<<<(unsigned)(BB * NC / 256), 256, 0, stream>>>(x, xT);

  // --- q^T [N,CQ], k^T [N,CQ], v [C,N]  (Bt = xT for all three) ---
  gemm_bf16<2><<<dim3(NN / NT, CQ / MT, BB), 256, 0, stream>>>(
      qwb, 0, C, xT, NC, C, qT, (size_t)NN * CQ, CQ, qb, nullptr, nullptr, 0, 0, C);
  gemm_bf16<2><<<dim3(NN / NT, CQ / MT, BB), 256, 0, stream>>>(
      kwb, 0, C, xT, NC, C, kT, (size_t)NN * CQ, CQ, kb, nullptr, nullptr, 0, 0, C);
  gemm_bf16<1><<<dim3(NN / NT, C / MT, BB), 256, 0, stream>>>(
      vwb, 0, C, xT, NC, C, vmat, CN, NN, vb, nullptr, nullptr, 0, 0, C);

  // --- scores = qT x kT^T -> att (bf16, L2-resident), softmax rows in place ---
  gemm_bf16<1><<<dim3(NN / NT, NN / MT, BB), 256, 0, stream>>>(
      qT, (size_t)NN * CQ, CQ, kT, (size_t)NN * CQ, CQ, att, NNl, NN,
      nullptr, nullptr, nullptr, 0, 0, CQ);
  softmax_rows<<<(unsigned)(BB * NN), 256, 0, stream>>>(att, NN);

  // --- catT[n, 0:C] = gamma*(att x v^T)[n,c] + x[c,n]  (fused epilogue) ---
  gemm_bf16<5><<<dim3(C / NT, NN / MT, BB), 256, 0, stream>>>(
      att, NNl, NN, vmat, CN, NN, catT, (size_t)NN * C2, C2,
      nullptr, gamma, x, CN, NN, NN);

  // --- energy = x x^T -> softmax -> catT[n, C:2C] = beta*(catt x x)[c,n] + x[c,n] ---
  gemm_bf16<1><<<dim3(C / NT, C / MT, BB), 256, 0, stream>>>(
      xbf, CN, NN, xbf, CN, NN, energy, (size_t)C * C, C,
      nullptr, nullptr, nullptr, 0, 0, NN);
  softmax_rows<<<(unsigned)(BB * C), 256, 0, stream>>>(energy, C);
  gemm_bf16<4><<<dim3(NN / NT, C / MT, BB), 256, 0, stream>>>(
      energy, (size_t)C * C, C, xT, NC, C, catT + C, (size_t)NN * C2, C2,
      nullptr, beta, x, CN, NN, C);

  // --- fusion conv: y = fw x catT^T + fb -> d_out (f32), then BN + ReLU ---
  gemm_bf16<0><<<dim3(NN / NT, C / MT, BB), 256, 0, stream>>>(
      fwb, 0, C2, catT, (size_t)NN * C2, C2, out, CN, NN,
      fb, nullptr, nullptr, 0, 0, C2);
  bn_stats<<<C, 256, 0, stream>>>(out, stats);
  bn_apply<<<(unsigned)(BB * CN / 256), 256, 0, stream>>>(out, stats, bn_w, bn_b);
}
